// RNNT_58136677318704
// MI455X (gfx1250) — matrix operand order repaired and checked by reading, not compiled, not run on hardware
//
#include <hip/hip_runtime.h>
#include <hip/hip_bf16.h>

typedef __attribute__((ext_vector_type(16))) _Float16 v16h;
typedef __attribute__((ext_vector_type(8)))  float    v8f;
typedef __attribute__((ext_vector_type(4)))  int      v4i;

#define B_    4
#define T_    400
#define U_    100
#define DJ    256
#define V_    1024
#define ROWS_X (B_*T_)   // 1600
#define ROWS_Y (B_*U_)   // 400

// LDS staging of packed W_cls: chunks of 4 N-tiles (4 * 8 KB = 32 KB), double
// buffered (64 KB LDS total). 64 N-tiles -> 16 chunks.
#define CHUNK_NB     4
#define NCHUNK       (64 / CHUNK_NB)                 // 16
#define CHUNK_HALFS  (CHUNK_NB * 8 * 512)            // 16384 halfs = 32 KB
#define CPY_ITERS    (CHUNK_HALFS * 2 / (256 * 16))  // 8 x b128 per thread

#define GLOBAL_AS __attribute__((address_space(1)))
#define LDS_AS    __attribute__((address_space(3)))

#if __has_builtin(__builtin_amdgcn_tanhf)
  #define FAST_TANH(x) __builtin_amdgcn_tanhf(x)
#elif __has_builtin(__builtin_amdgcn_tanh_f32)
  #define FAST_TANH(x) __builtin_amdgcn_tanh_f32(x)
#else
  #define FAST_TANH(x) tanhf(x)
#endif

#if __has_builtin(__builtin_amdgcn_global_load_async_to_lds_b128)
  #define HAVE_ASYNC_LDS 1
#else
  #define HAVE_ASYNC_LDS 0
#endif

// ---------------------------------------------------------------------------
// Projection: Out[row, j] = bias[j] + sum_k X[row,k] * W[k,j], j = 0..255.
// ---------------------------------------------------------------------------
__global__ void rnnt_proj_kernel(const float* __restrict__ X,
                                 const float* __restrict__ W,
                                 const float* __restrict__ bias,
                                 float* __restrict__ Out, int D)
{
    __shared__ float xs[768];
    const float* xrow = X + (size_t)blockIdx.x * D;
    for (int k = threadIdx.x; k < D; k += 256) xs[k] = xrow[k];
    __syncthreads();

    const int j = threadIdx.x;
    float acc = bias[j];
    for (int k = 0; k < D; ++k)
        acc = fmaf(xs[k], W[(size_t)k * DJ + j], acc);
    Out[(size_t)blockIdx.x * DJ + j] = acc;
}

// ---------------------------------------------------------------------------
// Pack W_cls (f32 [256,1024]) into f16 WMMA B-operand blocks (1 KB each):
//   block (nb, kb), stored at block_id = nb*8 + kb, layout [32 lanes][16 halfs]
//   lane L <  16 : column N = nb*16+L,      halfs = K kb*32 + 0..15
//   lane L >= 16 : column N = nb*16+(L-16), halfs = K kb*32 + 16..31
// ---------------------------------------------------------------------------
__global__ void rnnt_pack_wcls_kernel(const float* __restrict__ W,
                                      _Float16* __restrict__ Wc)
{
    const int blk  = blockIdx.x;          // 0..511
    const int nb   = blk >> 3;
    const int kb   = blk & 7;
    const int lane = threadIdx.x;         // 0..31
    const int n    = nb * 16 + (lane & 15);
    const int k0   = kb * 32 + ((lane & 16) ? 16 : 0);

    _Float16* dst = Wc + (size_t)blk * 512 + (size_t)lane * 16;
    #pragma unroll
    for (int j = 0; j < 16; ++j)
        dst[j] = (_Float16)W[(size_t)(k0 + j) * V_ + n];
}

// ---------------------------------------------------------------------------
// Copy one 32 KB chunk of packed W into an LDS buffer (256 threads cooperate).
// Async path: GLOBAL_LOAD_ASYNC_TO_LDS_B128 (ASYNCcnt). Fallback: b128 + ds.
// ---------------------------------------------------------------------------
__device__ __forceinline__ void copy_chunk(const _Float16* __restrict__ Wc,
                                           _Float16* __restrict__ smem,
                                           int chunk, int buf, int tid)
{
    const _Float16* src = Wc   + (size_t)chunk * CHUNK_HALFS;
    _Float16*       dst = smem + (size_t)buf   * CHUNK_HALFS;
    #pragma unroll
    for (int i = 0; i < CPY_ITERS; ++i) {
        const int off = (i * 256 + tid) * 8;        // halfs; 16 B per thread
#if HAVE_ASYNC_LDS
        __builtin_amdgcn_global_load_async_to_lds_b128(
            (GLOBAL_AS v4i*)(v4i*)(void*)(src + off),
            (LDS_AS    v4i*)(v4i*)(void*)(dst + off), 0, 0);
#else
        *(v4i*)(dst + off) = *(const v4i*)(src + off);
#endif
    }
}

__device__ __forceinline__ void wait_async_prev()
{   // older chunk complete; newer chunk (CPY_ITERS issues) may be in flight
#if HAVE_ASYNC_LDS
  #if __has_builtin(__builtin_amdgcn_s_wait_asynccnt)
    __builtin_amdgcn_s_wait_asynccnt(CPY_ITERS);
  #else
    asm volatile("s_wait_asynccnt 0x8" ::: "memory");
  #endif
#endif
}

__device__ __forceinline__ void wait_async_all()
{
#if HAVE_ASYNC_LDS
  #if __has_builtin(__builtin_amdgcn_s_wait_asynccnt)
    __builtin_amdgcn_s_wait_asynccnt(0);
  #else
    asm volatile("s_wait_asynccnt 0x0" ::: "memory");
  #endif
#endif
}

// ---------------------------------------------------------------------------
// Fused joint: per wave, A = tanh(xt[b,t,:] + yp[b,u,:]) for a 16-row m-tile
// (A-operand register layout), then sweep N-tiles with WMMA, B served from
// double-buffered LDS fed by async global->LDS DMA.
// ---------------------------------------------------------------------------
__global__ void __launch_bounds__(256)
rnnt_joint_kernel(const float* __restrict__ xt,
                  const float* __restrict__ yp,
                  const _Float16* __restrict__ Wc,
                  const float* __restrict__ b_cls,
                  float* __restrict__ out)
{
    __shared__ _Float16 smem[2 * CHUNK_HALFS];      // 64 KB

    const int tid   = threadIdx.x;
    const int wave  = tid >> 5;                     // 0..7
    const int lane  = tid & 31;
    const int mtile = blockIdx.x * 8 + wave;        // 0..9999
    const int m0    = mtile * 16;

    // Kick off chunk 0 DMA before doing the tanh work (overlap).
    copy_chunk(Wc, smem, 0, 0, tid);

    // This lane's matrix row within the tile.
    const int row = (lane & 15);
    const int m   = m0 + row;
    const int u   = m % U_;
    const int bt  = m / U_;                         // b*T + t
    const int b   = bt / T_;

    const float* xrow = xt + (size_t)bt * DJ;
    const float* yrow = yp + (size_t)(b * U_ + u) * DJ;

    // A-operand k order used here (contiguous, same as packed B):
    //   lane<16 : K = kb*32 + {0..15}
    //   lane>=16: K = kb*32 + {16..31}
    const int kbase = (lane < 16) ? 0 : 16;

    v16h a[8];
    #pragma unroll
    for (int kb = 0; kb < 8; ++kb) {
        const int k0 = kb * 32 + kbase;
        #pragma unroll
        for (int j = 0; j < 8; ++j) {
            float z0 = FAST_TANH(xrow[k0 + j]      + yrow[k0 + j]);
            float z1 = FAST_TANH(xrow[k0 + 8 + j] + yrow[k0 + 8 + j]);
            a[kb][j]     = (_Float16)z0;
            a[kb][8 + j] = (_Float16)z1;
        }
    }

    const int mrow = (lane < 16) ? 0 : 8;           // C/D layout M offset

    for (int c = 0; c < NCHUNK; ++c) {
        if (c + 1 < NCHUNK) {
            copy_chunk(Wc, smem, c + 1, (c + 1) & 1, tid);
            wait_async_prev();
        } else {
            wait_async_all();
        }
        __syncthreads();                            // chunk c visible to all

        const _Float16* buf = smem + (size_t)(c & 1) * CHUNK_HALFS;

        #pragma unroll
        for (int nbl = 0; nbl < CHUNK_NB; ++nbl) {
            v8f acc = {};
            #pragma unroll
            for (int kb = 0; kb < 8; ++kb) {
                v16h bm = *(const v16h*)(buf + (size_t)(nbl * 8 + kb) * 512
                                             + (size_t)lane * 16);
                acc = __builtin_amdgcn_wmma_f32_16x16x32_f16(
                          false, a[kb], false, bm,
                          (short)0, acc, false, false);
            }
            const int   n    = (c * CHUNK_NB + nbl) * 16 + (lane & 15);
            const float bias = b_cls[n];
            #pragma unroll
            for (int r = 0; r < 8; ++r)
                out[(size_t)(m0 + mrow + r) * V_ + n] = acc[r] + bias;
        }
        __syncthreads();                            // safe to overwrite buffer
    }
}

// ---------------------------------------------------------------------------
extern "C" void kernel_launch(void* const* d_in, const int* in_sizes, int n_in,
                              void* d_out, int out_size, void* d_ws, size_t ws_size,
                              hipStream_t stream)
{
    const float* x     = (const float*)d_in[0];
    const float* y     = (const float*)d_in[1];
    const float* W_tr  = (const float*)d_in[2];
    const float* b_tr  = (const float*)d_in[3];
    const float* W_pr  = (const float*)d_in[4];
    const float* b_pr  = (const float*)d_in[5];
    const float* W_cls = (const float*)d_in[6];
    const float* b_cls = (const float*)d_in[7];
    float* out = (float*)d_out;

    // Workspace: xt f32 [1600,256] | yp f32 [400,256] | Wc f16 [256*1024]
    float*    xt = (float*)d_ws;
    float*    yp = xt + (size_t)ROWS_X * DJ;
    _Float16* Wc = (_Float16*)(yp + (size_t)ROWS_Y * DJ);

    rnnt_proj_kernel<<<ROWS_X, 256, 0, stream>>>(x, W_tr, b_tr, xt, 768);
    rnnt_proj_kernel<<<ROWS_Y, 256, 0, stream>>>(y, W_pr, b_pr, yp, 640);
    rnnt_pack_wcls_kernel<<<512, 32, 0, stream>>>(W_cls, Wc);
    rnnt_joint_kernel<<<1250, 256, 0, stream>>>(xt, yp, Wc, b_cls, out);
}
